// GraphRelationLayer_38113539784771
// MI455X (gfx1250) — matrix operand order repaired and checked by reading, not compiled, not run on hardware
//
#include <hip/hip_runtime.h>
#include <hip/hip_bf16.h>

// Problem constants (match reference)
#define BB 8
#define NN 1024
#define DD 256
#define RR 4
#define BRN (BB*RR)   // 32

typedef _Float16 h2  __attribute__((ext_vector_type(2)));
typedef _Float16 h8  __attribute__((ext_vector_type(8)));
typedef _Float16 h16 __attribute__((ext_vector_type(16)));
typedef float    f8v __attribute__((ext_vector_type(8)));

// ---------------------------------------------------------------------------
// K0: convert node_features (f32) and rel_w (f32) to f16 scratch copies
// ---------------------------------------------------------------------------
__global__ void k_cvt(const float* __restrict__ X, const float* __restrict__ W,
                      _Float16* __restrict__ Xh, _Float16* __restrict__ Wh,
                      int nX, int nW) {
  int i = blockIdx.x * blockDim.x + threadIdx.x;
  if (i < nX) {
    Xh[i] = (_Float16)X[i];
  } else {
    int j = i - nX;
    if (j < nW) Wh[j] = (_Float16)W[j];
  }
}

// ---------------------------------------------------------------------------
// K1: feats = X @ W_r^T + b_r, stored transposed+f16: featsT[br][d][node]
// One wave computes a 16(node) x 64(d) tile; K-loop over D_IN in steps of 32.
// ---------------------------------------------------------------------------
__global__ __launch_bounds__(256) void k_gemm1(
    const _Float16* __restrict__ Xh, const _Float16* __restrict__ Wh,
    const float* __restrict__ rel_b, _Float16* __restrict__ fT) {
  const int wave = threadIdx.x >> 5, lane = threadIdx.x & 31;
  const int wid = blockIdx.x * 8 + wave;           // 0..8191
  const int dc = wid & 3;                          // d chunk (64 cols)
  const int it = (wid >> 2) & 63;                  // i tile (16 rows)
  const int r  = (wid >> 8) & 3;
  const int bb = wid >> 10;
  const int i0 = it * 16, d0 = dc * 64;
  const int n = lane & 15, g = lane >> 4;

  const _Float16* arow = Xh + (size_t)(bb * NN + i0 + n) * DD;

  f8v c[4];
#pragma unroll
  for (int t = 0; t < 4; ++t) c[t] = (f8v){0.f,0.f,0.f,0.f,0.f,0.f,0.f,0.f};

  for (int k0 = 0; k0 < DD; k0 += 32) {
    // A operand: row n, K = k0 + g*16..g*16+15 contiguous
    h8 alo = *(const h8*)(arow + k0 + g * 16);
    h8 ahi = *(const h8*)(arow + k0 + g * 16 + 8);
    h16 a = __builtin_shufflevector(alo, ahi, 0,1,2,3,4,5,6,7,8,9,10,11,12,13,14,15);
#pragma unroll
    for (int t = 0; t < 4; ++t) {
      // B operand: column d = d0+t*16+n of W^T == row of W; K = k0+g*16..+15 contiguous
      const _Float16* brow = Wh + (size_t)(r * DD + d0 + t * 16 + n) * DD;
      h16 bm = *(const h16*)(brow + k0 + g * 16);
      c[t] = __builtin_amdgcn_wmma_f32_16x16x32_f16(false, a, false, bm,
                                                    (short)0, c[t], false, false);
    }
  }

  const int br = bb * RR + r;
#pragma unroll
  for (int t = 0; t < 4; ++t) {
    const int dcol = d0 + t * 16 + n;
    const float bias = rel_b[r * DD + dcol];
    h8 hv;
#pragma unroll
    for (int v = 0; v < 8; ++v) hv[v] = (_Float16)(c[t][v] + bias);
    // C layout: lane col = dcol, rows i0 + g*8 + v (contiguous) -> b128 store
    *(h8*)(fT + (size_t)(br * DD + dcol) * NN + i0 + g * 8) = hv;
  }
}

// ---------------------------------------------------------------------------
// K2: si[br,i] = feats . attn_w[:D],  sj[br,i] = feats . attn_w[D:]
// ---------------------------------------------------------------------------
__global__ __launch_bounds__(256) void k_sisj(
    const _Float16* __restrict__ fT, const float* __restrict__ attn_w,
    float* __restrict__ si, float* __restrict__ sj) {
  int idx = blockIdx.x * 256 + threadIdx.x;        // 0..32767
  int br = idx >> 10, i = idx & 1023;
  float a = 0.f, bsum = 0.f;
  for (int d = 0; d < DD; ++d) {
    float f = (float)fT[(size_t)(br * DD + d) * NN + i];
    a    += f * attn_w[d];
    bsum += f * attn_w[DD + d];
  }
  si[idx] = a;
  sj[idx] = bsum;
}

// ---------------------------------------------------------------------------
// K3: per-row softmax statistics (max, 1/sum) streaming adj. Wave per row.
// ---------------------------------------------------------------------------
__global__ __launch_bounds__(256) void k_rowstat(
    const int* __restrict__ adj, const float* __restrict__ si,
    const float* __restrict__ sj, const float* __restrict__ attn_b,
    float* __restrict__ mrow, float* __restrict__ srow) {
  const int wave = threadIdx.x >> 5, lane = threadIdx.x & 31;
  const int row = blockIdx.x * 8 + wave;           // 0..32767
  const int br = row >> 10, i = row & 1023;
  const float c = si[row] + attn_b[0];
  const int* arow = adj + ((size_t)br << 20) + (size_t)i * NN;
  const float* sjr = sj + (br << 10);

  float m = -3.0e38f, s = 0.f;
  for (int t = 0; t < 32; ++t) {
    int j = lane + t * 32;
    float x = arow[j] ? (c + sjr[j]) : -1e9f;
    float nm = fmaxf(m, x);
    s = s * __expf(m - nm) + __expf(x - nm);
    m = nm;
  }
#pragma unroll
  for (int off = 16; off > 0; off >>= 1) {
    float om = __shfl_xor(m, off, 32);
    float os = __shfl_xor(s, off, 32);
    float nm = fmaxf(m, om);
    s = s * __expf(m - nm) + os * __expf(om - nm);
    m = nm;
  }
  if (lane == 0) { mrow[row] = m; srow[row] = 1.0f / s; }
}

// ---------------------------------------------------------------------------
// K4: fused attention-weighted aggregation + gating.
// Block = 256 threads (8 waves) per (b, 16-row i-tile).
// Cooperative exp tile (16x32) -> LDS; each wave owns 32 d-columns (2 WMMA
// tiles), accumulates C over r and all j tiles; LDS-staged gate epilogue.
// ---------------------------------------------------------------------------
__global__ __launch_bounds__(256) void k_attn_agg(
    const int* __restrict__ adj, const _Float16* __restrict__ fT,
    const float* __restrict__ si, const float* __restrict__ sj,
    const float* __restrict__ mrow, const float* __restrict__ srow,
    const float* __restrict__ attn_b, const float* __restrict__ gate_w,
    const float* __restrict__ gate_b, float* __restrict__ out) {
  __shared__ _Float16 aT[16 * 32];       // attention A tile (f16)
  __shared__ float scL[16], mrL[16], srL[16];
  __shared__ float aggL[16 * 256];       // f32 aggregate staging
  __shared__ float gac[16];
  __shared__ float gateL[16];

  const int t = threadIdx.x;
  const int wave = t >> 5, lane = t & 31;
  const int b = blockIdx.x >> 6, it = blockIdx.x & 63;
  const int i0 = it * 16;
  const int n = lane & 15, g = lane >> 4;
  const int em = t >> 4;                 // exp-tile row for this thread
  const int ek = (t * 2) & 31;           // exp-tile col pair start
  const float ab = attn_b[0];

  f8v c0 = (f8v){0.f,0.f,0.f,0.f,0.f,0.f,0.f,0.f};
  f8v c1 = c0;

  for (int r = 0; r < RR; ++r) {
    const int br = b * RR + r;
    if (t < 16) {
      int rowid = (br << 10) + i0 + t;
      scL[t] = si[rowid] + ab;
      mrL[t] = mrow[rowid];
      srL[t] = srow[rowid];             // already 1/sum
    }
    __syncthreads();

    const int*      abase = adj + ((size_t)br << 20) + (size_t)(i0 + em) * NN + ek;
    const float*    sjb   = sj + (br << 10) + ek;
    const _Float16* fb0   = fT + (size_t)(br * DD + wave * 32 + n) * NN;
    const _Float16* fb1   = fb0 + (size_t)16 * NN;

    for (int jt = 0; jt < 32; ++jt) {
      const int j0 = jt * 32;
      // --- cooperative exp tile: 2 elements per thread ---
      int2   a2 = *(const int2*)(abase + j0);
      float2 s2 = *(const float2*)(sjb + j0);
      float cc = scL[em], mm = mrL[em], iv = srL[em];
      float x0 = a2.x ? (cc + s2.x) : -1e9f;
      float x1 = a2.y ? (cc + s2.y) : -1e9f;
      h2 pk = { (_Float16)(__expf(x0 - mm) * iv),
                (_Float16)(__expf(x1 - mm) * iv) };
      *(h2*)&aT[em * 32 + ek] = pk;
      __syncthreads();

      // --- A operand from LDS (row n, K contiguous) ---
      h8 alo = *(const h8*)&aT[n * 32 + g * 16];
      h8 ahi = *(const h8*)&aT[n * 32 + g * 16 + 8];
      h16 a = __builtin_shufflevector(alo, ahi, 0,1,2,3,4,5,6,7,8,9,10,11,12,13,14,15);
      // --- B operands from d-major featsT (K contiguous) ---
      h16 b0 = *(const h16*)(fb0 + j0 + g * 16);
      h16 b1 = *(const h16*)(fb1 + j0 + g * 16);
      c0 = __builtin_amdgcn_wmma_f32_16x16x32_f16(false, a, false, b0, (short)0, c0, false, false);
      c1 = __builtin_amdgcn_wmma_f32_16x16x32_f16(false, a, false, b1, (short)0, c1, false, false);
      __syncthreads();
    }
  }

  // stage aggregate (lane col = wave*32+n / +16; rows g*8+v)
#pragma unroll
  for (int v = 0; v < 8; ++v) {
    aggL[(g * 8 + v) * 256 + wave * 32 + n]      = c0[v];
    aggL[(g * 8 + v) * 256 + wave * 32 + 16 + n] = c1[v];
  }
  __syncthreads();

  if (t < 16) gac[t] = 0.f;
  __syncthreads();
  {
    const int m = t >> 4, q = t & 15;
    float p = 0.f;
#pragma unroll
    for (int k = 0; k < 16; ++k) p += aggL[m * 256 + q * 16 + k] * gate_w[q * 16 + k];
    atomicAdd(&gac[m], p);
  }
  __syncthreads();
  if (t < 16) gateL[t] = 1.0f / (1.0f + __expf(-(gac[t] + gate_b[0])));
  __syncthreads();
  {
    const int m = t >> 4, q = t & 15;
    const float gt = gateL[m];
    float* orow = out + (size_t)(b * NN + i0 + m) * DD + q * 16;
#pragma unroll
    for (int k = 0; k < 4; ++k) {
      float4 v4 = *(float4*)&aggL[m * 256 + q * 16 + k * 4];
      float4 w4 = make_float4(v4.x * gt, v4.y * gt, v4.z * gt, v4.w * gt);
      *(float4*)(orow + k * 4) = w4;
    }
  }
}

// ---------------------------------------------------------------------------
// Host-side launch
// ---------------------------------------------------------------------------
extern "C" void kernel_launch(void* const* d_in, const int* in_sizes, int n_in,
                              void* d_out, int out_size, void* d_ws, size_t ws_size,
                              hipStream_t stream) {
  const float* X      = (const float*)d_in[0];   // [B,N,D_IN]
  const int*   adj    = (const int*)  d_in[1];   // [B,R,N,N]
  const float* rel_w  = (const float*)d_in[2];   // [R,D_OUT,D_IN]
  const float* rel_b  = (const float*)d_in[3];   // [R,D_OUT]
  const float* attn_w = (const float*)d_in[4];   // [2*D_OUT]
  const float* attn_b = (const float*)d_in[5];   // scalar
  const float* gate_w = (const float*)d_in[6];   // [D_OUT]
  const float* gate_b = (const float*)d_in[7];   // scalar
  float* out = (float*)d_out;

  // workspace layout (f16-heavy, ~21 MB total)
  _Float16* Xh = (_Float16*)d_ws;                         // B*N*D
  _Float16* Wh = Xh + (size_t)BB * NN * DD;               // R*D*D
  _Float16* fT = Wh + (size_t)RR * DD * DD;               // B*R*D*N (d-major)
  float* si   = (float*)(fT + (size_t)BB * RR * DD * NN); // BR*N
  float* sj   = si + BRN * NN;
  float* mrow = sj + BRN * NN;
  float* srow = mrow + BRN * NN;

  const int nX = BB * NN * DD;       // 2,097,152
  const int nW = RR * DD * DD;       // 262,144

  k_cvt<<<(nX + nW + 255) / 256, 256, 0, stream>>>(X, rel_w, Xh, Wh, nX, nW);
  k_gemm1<<<1024, 256, 0, stream>>>(Xh, Wh, rel_b, fT);
  k_sisj<<<(BRN * NN) / 256, 256, 0, stream>>>(fT, attn_w, si, sj);
  k_rowstat<<<(BRN * NN) / 8, 256, 0, stream>>>(adj, si, sj, attn_b, mrow, srow);
  k_attn_agg<<<BB * 64, 256, 0, stream>>>(adj, fT, si, sj, mrow, srow,
                                          attn_b, gate_w, gate_b, out);
}
